// PointNet_39599598469651
// MI455X (gfx1250) — compile-verified, matrix operand order checked by reading, not run on hardware
//
#include <hip/hip_runtime.h>
#include <hip/hip_bf16.h>

typedef __attribute__((ext_vector_type(16))) _Float16 v16h;
typedef __attribute__((ext_vector_type(8)))  _Float16 v8h;
typedef __attribute__((ext_vector_type(8)))  float    v8f;

#define NB 32
#define NP 2048
#define KN 16
#define NH 128

// ---------------------------------------------------------------------------
// Weight pre-swizzle: f32 row-major W[k][n] (n<128) -> f16 in WMMA-B operand
// order, so each wave's B tile for (ks, t) is one contiguous 1KB chunk (32B per lane):
//   out[ ((ks*8 + t)*32 + lane)*16 + h ],  lane = hi*16 + (n%16), t = n/16
//   kin(h,hi) = (h<8) ? hi*8+h : 16 + hi*8 + (h-8),  k = ks*32 + kin
// Rows >= `rows` are zero padding.
// ---------------------------------------------------------------------------
__global__ void prep_w_kernel(const float* __restrict__ W, int rows, int nk,
                              _Float16* __restrict__ out) {
  int o = blockIdx.x * 256 + threadIdx.x;
  if (o >= nk * 4096) return;
  int ks = o >> 12, rem = o & 4095;
  int t = rem >> 9, lane = (rem >> 4) & 31, h = rem & 15;
  int hi = lane >> 4, nlo = lane & 15;
  int n = t * 16 + nlo;
  int kin = (h < 8) ? (hi * 8 + h) : (16 + hi * 8 + (h - 8));
  int k = ks * 32 + kin;
  float v = (k < rows) ? W[k * NH + n] : 0.0f;
  out[o] = (_Float16)v;
}

// ---------------------------------------------------------------------------
// KNN: one thread per query point, unsorted top-16 with tracked worst slot.
// Candidates streamed through LDS as float4 (x,y,z,|p|^2).
// ---------------------------------------------------------------------------
__global__ __launch_bounds__(256) void knn_kernel(const float* __restrict__ pos,
                                                  int* __restrict__ nbr) {
  const int b = blockIdx.x >> 3;
  const int q = ((blockIdx.x & 7) << 8) + threadIdx.x;
  const float* pb = pos + (size_t)b * NP * 3;
  const float qx = pb[q * 3 + 0], qy = pb[q * 3 + 1], qz = pb[q * 3 + 2];
  const float qs = qx * qx + qy * qy + qz * qz;
  __shared__ float4 sp[256];
  float bd[KN];
  int   bi[KN];
#pragma unroll
  for (int s = 0; s < KN; ++s) { bd[s] = 3.4e38f; bi[s] = 0; }
  float worst = 3.4e38f;
  int wslot = 0;
  for (int tile = 0; tile < NP; tile += 256) {
    __syncthreads();
    int c = tile + threadIdx.x;
    float cx = pb[c * 3 + 0], cy = pb[c * 3 + 1], cz = pb[c * 3 + 2];
    sp[threadIdx.x] = make_float4(cx, cy, cz, cx * cx + cy * cy + cz * cz);
    __syncthreads();
    for (int j = 0; j < 256; ++j) {
      float4 cc = sp[j];
      float d2 = qs + cc.w - 2.0f * (qx * cc.x + qy * cc.y + qz * cc.z);
      if (d2 < worst) {
        bd[wslot] = d2;
        bi[wslot] = tile + j;
        worst = bd[0];
        wslot = 0;
#pragma unroll
        for (int s = 1; s < KN; ++s)
          if (bd[s] > worst) { worst = bd[s]; wslot = s; }
      }
    }
  }
  int* o = nbr + ((size_t)b * NP + q) * KN;
#pragma unroll
  for (int s = 0; s < KN; ++s) o[s] = bi[s];
}

__device__ __forceinline__ v16h cat16(v8h lo, v8h h8) {
  return __builtin_shufflevector(lo, h8, 0, 1, 2, 3, 4, 5, 6, 7,
                                 8, 9, 10, 11, 12, 13, 14, 15);
}

// ---------------------------------------------------------------------------
// One PointNet layer. One wave = one point (16 neighbor messages = 16-row
// WMMA A tile). Stage1: [16,KA]x[KA,128]; ReLU; restage via LDS into A layout;
// Stage2: [16,128]x[128,128]; bias; max over 16 rows; ReLU; store.
// ---------------------------------------------------------------------------
template <int CIN, int KA>
__global__ __launch_bounds__(256) void layer_kernel(
    const float* __restrict__ hin, const float* __restrict__ pos,
    const int* __restrict__ nbr,
    const _Float16* __restrict__ WaSw, const float* __restrict__ ba,
    const _Float16* __restrict__ WbSw, const float* __restrict__ bb,
    float* __restrict__ hout) {
  constexpr int NK1 = KA / 32;
  constexpr int SB = 16 * (KA > NH ? KA : NH);
  __shared__ _Float16 sBuf[8][SB];
  __shared__ int sIdx[8][16];
  const int wave = threadIdx.x >> 5;
  const int lane = threadIdx.x & 31;
  const int p = blockIdx.x * 8 + wave;          // global point id
  const int b = p / NP;
  const size_t bRow = (size_t)b * NP;
  _Float16* msg = sBuf[wave];

  if (lane < 16) sIdx[wave][lane] = nbr[(size_t)p * KN + lane];
  __syncthreads();

  // Gather messages [16 x KA]: cols [0,CIN)=h_j, [CIN,CIN+3)=pos_j-pos_i, rest 0
  for (int idx = lane; idx < 16 * KA; idx += 32) {
    int r = idx / KA, c = idx - r * KA;
    int nj = sIdx[wave][r];
    size_t row = bRow + (size_t)nj;
    float v = 0.0f;
    if (c < CIN) {
      v = hin[row * CIN + c];
    } else if (c < CIN + 3) {
      int d = c - CIN;
      v = pos[row * 3 + d] - pos[(size_t)p * 3 + d];
    }
    msg[idx] = (_Float16)v;
  }
  __syncthreads();

  const int m = lane & 15, hi = lane >> 4;
  const v16h* Wa16 = (const v16h*)WaSw;
  const v16h* Wb16 = (const v16h*)WbSw;

  // ---- stage 1: msg[16,KA] x Wa[KA,128] ----
  v8f acc[8];
#pragma unroll
  for (int t = 0; t < 8; ++t)
#pragma unroll
    for (int r = 0; r < 8; ++r) acc[t][r] = 0.0f;

#pragma unroll
  for (int ks = 0; ks < NK1; ++ks) {
    v8h lo = *(const v8h*)(msg + m * KA + ks * 32 + hi * 8);
    v8h h8 = *(const v8h*)(msg + m * KA + ks * 32 + 16 + hi * 8);
    v16h A = cat16(lo, h8);
#pragma unroll
    for (int t = 0; t < 8; ++t) {
      v16h Bv = Wa16[(ks * 8 + t) * 32 + lane];
      acc[t] = __builtin_amdgcn_wmma_f32_16x16x32_f16(
          false, A, false, Bv, (short)0, acc[t], false, false);
    }
  }
  __syncthreads();

  // bias + ReLU, restage intermediate [16,128] f16 row-major (reuse msg buf)
#pragma unroll
  for (int t = 0; t < 8; ++t) {
    float bias = ba[t * 16 + m];
#pragma unroll
    for (int r = 0; r < 8; ++r) {
      float v = acc[t][r] + bias;
      v = v > 0.0f ? v : 0.0f;
      msg[(r + 8 * hi) * NH + t * 16 + m] = (_Float16)v;
    }
  }
  __syncthreads();

  // ---- stage 2: inter[16,128] x Wb[128,128] ----
  v8f acc2[8];
#pragma unroll
  for (int t = 0; t < 8; ++t)
#pragma unroll
    for (int r = 0; r < 8; ++r) acc2[t][r] = 0.0f;

#pragma unroll
  for (int ks = 0; ks < 4; ++ks) {
    v8h lo = *(const v8h*)(msg + m * NH + ks * 32 + hi * 8);
    v8h h8 = *(const v8h*)(msg + m * NH + ks * 32 + 16 + hi * 8);
    v16h A = cat16(lo, h8);
#pragma unroll
    for (int t = 0; t < 8; ++t) {
      v16h Bv = Wb16[(ks * 8 + t) * 32 + lane];
      acc2[t] = __builtin_amdgcn_wmma_f32_16x16x32_f16(
          false, A, false, Bv, (short)0, acc2[t], false, false);
    }
  }

  // bias + max over 16 rows (8 in-lane + shfl_xor(16)) + ReLU, store
#pragma unroll
  for (int t = 0; t < 8; ++t) {
    float bias = bb[t * 16 + m];
    float mx = -3.4e38f;
#pragma unroll
    for (int r = 0; r < 8; ++r) mx = fmaxf(mx, acc2[t][r] + bias);
    float other = __shfl_xor(mx, 16, 32);
    mx = fmaxf(mx, other);
    mx = mx > 0.0f ? mx : 0.0f;
    if (hi == 0) hout[(size_t)p * NH + t * 16 + m] = mx;
  }
}

// ---------------------------------------------------------------------------
// Global max pool over P, then [32,128]x[128,10] classifier.
// ---------------------------------------------------------------------------
__global__ __launch_bounds__(128) void classifier_kernel(
    const float* __restrict__ h2, const float* __restrict__ Wc,
    const float* __restrict__ bc, float* __restrict__ out) {
  const int b = blockIdx.x, c = threadIdx.x;
  const float* hb = h2 + (size_t)b * NP * NH;
  float mx = -3.4e38f;
  for (int j = 0; j < NP; ++j) mx = fmaxf(mx, hb[(size_t)j * NH + c]);
  __shared__ float g[NH];
  g[c] = mx;
  __syncthreads();
  if (c < 10) {
    float s = bc[c];
#pragma unroll 8
    for (int k = 0; k < NH; ++k) s += g[k] * Wc[k * 10 + c];
    out[b * 10 + c] = s;
  }
}

// ---------------------------------------------------------------------------
extern "C" void kernel_launch(void* const* d_in, const int* in_sizes, int n_in,
                              void* d_out, int out_size, void* d_ws,
                              size_t ws_size, hipStream_t stream) {
  const float* pos = (const float*)d_in[0];
  // d_in[1] = batch (unused: graphs are equal-size, sorted)
  const float* W1a = (const float*)d_in[2];
  const float* b1a = (const float*)d_in[3];
  const float* W1b = (const float*)d_in[4];
  const float* b1b = (const float*)d_in[5];
  const float* W2a = (const float*)d_in[6];
  const float* b2a = (const float*)d_in[7];
  const float* W2b = (const float*)d_in[8];
  const float* b2b = (const float*)d_in[9];
  const float* Wc  = (const float*)d_in[10];
  const float* bc  = (const float*)d_in[11];
  float* out = (float*)d_out;

  char* ws = (char*)d_ws;
  size_t off = 0;
  auto alloc = [&](size_t bytes) -> void* {
    void* r = ws + off;
    off = (off + bytes + 255) & ~(size_t)255;
    return r;
  };
  int*      nbr = (int*)alloc((size_t)NB * NP * KN * sizeof(int));
  float*    h1  = (float*)alloc((size_t)NB * NP * NH * sizeof(float));
  float*    h2  = (float*)alloc((size_t)NB * NP * NH * sizeof(float));
  _Float16* w1a = (_Float16*)alloc((size_t)1 * 4096 * sizeof(_Float16));
  _Float16* w1b = (_Float16*)alloc((size_t)4 * 4096 * sizeof(_Float16));
  _Float16* w2a = (_Float16*)alloc((size_t)5 * 4096 * sizeof(_Float16));
  _Float16* w2b = (_Float16*)alloc((size_t)4 * 4096 * sizeof(_Float16));

  // weight conversion + swizzle to B-operand order (f32 -> f16)
  prep_w_kernel<<<(1 * 4096 + 255) / 256, 256, 0, stream>>>(W1a, 6, 1, w1a);
  prep_w_kernel<<<(4 * 4096 + 255) / 256, 256, 0, stream>>>(W1b, NH, 4, w1b);
  prep_w_kernel<<<(5 * 4096 + 255) / 256, 256, 0, stream>>>(W2a, NH + 3, 5, w2a);
  prep_w_kernel<<<(4 * 4096 + 255) / 256, 256, 0, stream>>>(W2b, NH, 4, w2b);

  // KNN (top-16 incl. self, order-free since aggregation is max)
  knn_kernel<<<NB * 8, 256, 0, stream>>>(pos, nbr);

  // layer 1: message = [pos_j(3), rel(3)] -> pad K to 32
  layer_kernel<3, 32><<<NB * NP / 8, 256, 0, stream>>>(
      pos, pos, nbr, w1a, b1a, w1b, b1b, h1);
  // layer 2: message = [h_j(128), rel(3)] -> pad K to 160
  layer_kernel<128, 160><<<NB * NP / 8, 256, 0, stream>>>(
      h1, pos, nbr, w2a, b2a, w2b, b2b, h2);

  classifier_kernel<<<NB, 128, 0, stream>>>(h2, Wc, bc, out);
}
